// Actor_77893526880605
// MI455X (gfx1250) — compile-verified, matrix operand order checked by reading, not run on hardware
//
#include <hip/hip_runtime.h>
#include <hip/hip_bf16.h>
#include <cstdint>

#define BB   128
#define TSEQ 1024
#define TOT  1025
#define FF   64
#define DD   128
#define HH   256
#define GG   768   // 3*H
#define KW   384   // packed [Wih|Whh] columns
#define WDIM 128
#define MAXS 8
#define NBEAM 4

typedef __attribute__((ext_vector_type(16))) _Float16 v16h;
typedef __attribute__((ext_vector_type(8)))  float    v8f;
typedef __attribute__((ext_vector_type(4)))  unsigned int v4u;
typedef __attribute__((ext_vector_type(4)))  int      v4i;
typedef __attribute__((ext_vector_type(8)))  int      v8i;

#if __has_builtin(__builtin_amdgcn_tensor_load_to_lds)
#define USE_TDM 1
#else
#define USE_TDM 0
#endif

// ---------------------------------------------------------------- helpers ---

__device__ __forceinline__ v8f wmma_f16(v16h a, v16h b, v8f c) {
  // D = A(16x32 f16) x B(32x16 f16) + C(16x16 f32)
  return __builtin_amdgcn_wmma_f32_16x16x32_f16(false, a, false, b, (short)0, c,
                                                false, false);
}

// Load a 16x32 f16 fragment from row-major memory (row stride ld elements).
// Per ISA 7.12.2: lanes 0-15 hold rows M=0..15, halves 0..7 = K 0..7 and
// halves 8..15 = K 16..23; lanes 16-31 hold K 8..15 / 24..31.
__device__ __forceinline__ v16h load_frag(const _Float16* __restrict__ p,
                                          int ld, int kOff) {
  const int l = (int)(threadIdx.x & 31u);
  const _Float16* r = p + (l & 15) * ld + kOff + ((l >> 4) << 3);
  v16h f;
#pragma unroll
  for (int i = 0; i < 8; ++i) f[i] = r[i];
#pragma unroll
  for (int i = 0; i < 8; ++i) f[8 + i] = r[16 + i];
  return f;
}

// branch-free transcendentals: raw v_rcp_f32 / native v_tanh_f32 when exposed
__device__ __forceinline__ float fast_rcp(float x) {
#if __has_builtin(__builtin_amdgcn_rcpf)
  return __builtin_amdgcn_rcpf(x);
#else
  return 1.f / x;
#endif
}
__device__ __forceinline__ float fast_tanh(float x) {
#if __has_builtin(__builtin_amdgcn_tanhf)
  return __builtin_amdgcn_tanhf(x);
#elif __has_builtin(__builtin_amdgcn_tanh_f32)
  return __builtin_amdgcn_tanh_f32(x);
#else
  float ax = fabsf(x);
  float e  = __expf(-2.f * ax);
  float t  = (1.f - e) * fast_rcp(1.f + e);
  return copysignf(t, x);
#endif
}
__device__ __forceinline__ float sigm(float x) {
  return fast_rcp(1.f + __expf(-x));
}

#if USE_TDM
// TDM: load a 16-row x DD-col f16 tile (row stride = TOT*DD elements) from
// global into LDS (contiguous 16*DD halves). D# per ISA 8.3/8.4.
__device__ __forceinline__ void tdm_load_x(unsigned long long gaddr,
                                           unsigned int lds_addr) {
  v4u g0 = {1u,                              // count=1 (valid user D#)
            lds_addr,                        // lds_addr (bytes)
            (unsigned int)gaddr,             // global_addr[31:0]
            (unsigned int)((gaddr >> 32) & 0x1FFFFFFu) | (2u << 30)}; // [56:32] | type=2
  v8i g1;
  g1[0] = 0x00010000;                        // wg_mask=0, data_size=1 (2B)
  g1[1] = (int)(((unsigned)DD & 0xFFFFu) << 16);   // tensor_dim0[15:0]
  g1[2] = (int)((16u & 0xFFFFu) << 16);            // tensor_dim1[15:0] = 16 rows
  g1[3] = (int)(((unsigned)DD & 0xFFFFu) << 16);   // tile_dim0 = DD
  g1[4] = 16;                                      // tile_dim1 = 16 (tile_dim2=0)
  g1[5] = (int)(TOT * DD);                         // tensor_dim0_stride[31:0]
  g1[6] = 0;                                       // stride[47:32]=0, dim1_stride lo=0
  g1[7] = 0;
  v4i gz = {0, 0, 0, 0};
#if defined(__clang_major__) && __clang_major__ >= 23
  v8i gz8 = {0, 0, 0, 0, 0, 0, 0, 0};
  __builtin_amdgcn_tensor_load_to_lds(g0, g1, gz, gz, gz8, 0);
#else
  __builtin_amdgcn_tensor_load_to_lds(g0, g1, gz, gz, 0);
#endif
}
#endif

// ------------------------------------------------------------ weight prep ---

__global__ void cvt_f16(_Float16* __restrict__ dst, const float* __restrict__ src, int n) {
  for (int i = blockIdx.x * blockDim.x + threadIdx.x; i < n; i += gridDim.x * blockDim.x)
    dst[i] = (_Float16)src[i];
}

// dst[768][384] = [ Wih(768x128) | Whh(768x256) ]
__global__ void pack_gru(_Float16* __restrict__ dst, const float* __restrict__ wih,
                         const float* __restrict__ whh) {
  for (int i = blockIdx.x * blockDim.x + threadIdx.x; i < GG * KW;
       i += gridDim.x * blockDim.x) {
    int row = i / KW, col = i % KW;
    float v = (col < DD) ? wih[row * DD + col] : whh[row * HH + (col - DD)];
    dst[i] = (_Float16)v;
  }
}

__global__ void init_bos(const _Float16* __restrict__ emb, _Float16* __restrict__ enc_in) {
  int i = blockIdx.x * blockDim.x + threadIdx.x;           // BB*DD
  if (i < BB * DD)
    enc_in[(long)(i >> 7) * TOT * DD + (i & 127)] = emb[i & 127];
}

// ------------------------------------------------- input projection (WMMA) ---
// enc_in[b][t+1][:] = relu(tanh(1e6*(cm[b][t] @ linW^T) + lb))
__global__ __launch_bounds__(256) void proj_kernel(
    const float* __restrict__ cm, const _Float16* __restrict__ Wl,
    const float* __restrict__ lb, _Float16* __restrict__ enc_in) {
  __shared__ _Float16 at[16][FF + 8];
  const int wave = threadIdx.x >> 5, lane = threadIdx.x & 31;
  const int n0 = wave * 16;
  for (int mt = blockIdx.x; mt < (BB * TSEQ) / 16; mt += gridDim.x) {
    const int mrow = mt * 16;
    for (int i = threadIdx.x; i < 16 * FF; i += blockDim.x)
      at[i >> 6][i & 63] = (_Float16)cm[(long)mrow * FF + i];
    __syncthreads();
    v8f acc = {0.f, 0.f, 0.f, 0.f, 0.f, 0.f, 0.f, 0.f};
    acc = wmma_f16(load_frag(&at[0][0], FF + 8, 0),  load_frag(Wl + n0 * FF, FF, 0),  acc);
    acc = wmma_f16(load_frag(&at[0][0], FF + 8, 32), load_frag(Wl + n0 * FF, FF, 32), acc);
    const int n = lane & 15, mo = (lane >> 4) * 8;
#pragma unroll
    for (int r = 0; r < 8; ++r) {
      int m = mrow + mo + r;
      int b = m >> 10, tt = m & 1023;
      float v = fast_tanh(1e6f * acc[r] + lb[n0 + n]);
      v = fmaxf(v, 0.f);
      enc_in[((long)b * TOT + tt + 1) * DD + n0 + n] = (_Float16)v;
    }
    __syncthreads();
  }
}

// ----------------------------------------------- persistent encoder GRU scan -
// grid = 8 blocks x 512 threads; block owns 16 batch rows, loops t=0..1024.
// x tile staged into LDS by the Tensor Data Mover (double-buffered).
__global__ __launch_bounds__(512) void encoder_scan(
    const _Float16* __restrict__ enc_in, const _Float16* __restrict__ Wp,
    const float* __restrict__ bih, const float* __restrict__ bhh,
    _Float16* __restrict__ enc_out) {
  __shared__ _Float16 hbuf[16][HH + 8];
  __shared__ float grz[16][512];
  __shared__ float gin[16][HH];
  __shared__ float ghn[16][HH];
#if USE_TDM
  __shared__ _Float16 xstage[2][16 * DD];
#endif
  const int m0 = blockIdx.x * 16;
  const int wave = threadIdx.x >> 5, lane = threadIdx.x & 31;

  for (int i = threadIdx.x; i < 16 * HH; i += blockDim.x)
    hbuf[i >> 8][i & 255] = (_Float16)0.f;

#if USE_TDM
  if (threadIdx.x < 32) {   // wave 0 drives the TDM (EXEC ignored by TDM)
    tdm_load_x((unsigned long long)(uintptr_t)(enc_in + (long)m0 * TOT * DD),
               (unsigned int)(uintptr_t)&xstage[0][0]);
    __builtin_amdgcn_s_wait_tensorcnt(0);
  }
#endif
  __syncthreads();

  for (int t = 0; t < TOT; ++t) {
#if USE_TDM
    if (threadIdx.x < 32 && t + 1 < TOT)   // prefetch next x tile via TDM
      tdm_load_x((unsigned long long)(uintptr_t)
                     (enc_in + ((long)m0 * TOT + t + 1) * DD),
                 (unsigned int)(uintptr_t)&xstage[(t + 1) & 1][0]);
    const _Float16* xbase = &xstage[t & 1][0];
    const int xld = DD;
#else
    const _Float16* xbase = enc_in + ((long)m0 * TOT + t) * DD;
    const int xld = TOT * DD;
    if (t + 1 < TOT && threadIdx.x < 16)
      __builtin_prefetch(enc_in + ((long)(m0 + threadIdx.x) * TOT + t + 1) * DD, 0, 1);
#endif
    v16h ax[4];
#pragma unroll
    for (int kk = 0; kk < 4; ++kk) ax[kk] = load_frag(xbase, xld, kk * 32);
    v16h ah[8];
#pragma unroll
    for (int kk = 0; kk < 8; ++kk) ah[kk] = load_frag(&hbuf[0][0], HH + 8, kk * 32);

#pragma unroll
    for (int i = 0; i < 3; ++i) {
      const int nt = wave * 3 + i;           // 0..47
      const int n0 = nt * 16;
      const _Float16* wb = Wp + (long)n0 * KW;
      v8f accI = {0.f, 0.f, 0.f, 0.f, 0.f, 0.f, 0.f, 0.f};
      v8f accH = {0.f, 0.f, 0.f, 0.f, 0.f, 0.f, 0.f, 0.f};
#pragma unroll
      for (int kk = 0; kk < 4; ++kk)
        accI = wmma_f16(ax[kk], load_frag(wb, KW, kk * 32), accI);
#pragma unroll
      for (int kk = 0; kk < 8; ++kk)
        accH = wmma_f16(ah[kk], load_frag(wb, KW, DD + kk * 32), accH);
      const int n = lane & 15, mo = (lane >> 4) * 8;
      if (n0 < 512) {
#pragma unroll
        for (int r = 0; r < 8; ++r) grz[mo + r][n0 + n] = accI[r] + accH[r];
      } else {
#pragma unroll
        for (int r = 0; r < 8; ++r) {
          gin[mo + r][n0 - 512 + n] = accI[r];
          ghn[mo + r][n0 - 512 + n] = accH[r];
        }
      }
    }
    __syncthreads();

    for (int i = threadIdx.x; i < 16 * HH; i += blockDim.x) {
      const int mm = i >> 8, j = i & 255;
      float r = sigm(grz[mm][j] + bih[j] + bhh[j]);
      float z = sigm(grz[mm][256 + j] + bih[256 + j] + bhh[256 + j]);
      float nv = fast_tanh(gin[mm][j] + bih[512 + j] + r * (ghn[mm][j] + bhh[512 + j]));
      float hp = (float)hbuf[mm][j];
      float hn = (1.f - z) * nv + z * hp;
      hbuf[mm][j] = (_Float16)hn;
      enc_out[((long)(m0 + mm) * TOT + t) * HH + j] = (_Float16)hn;
    }
#if USE_TDM
    if (threadIdx.x < 32) __builtin_amdgcn_s_wait_tensorcnt(0);
#endif
    __syncthreads();
  }
}

// ------------------------------------------------------- Key = enc_out @ Wk^T
// KeyT[b][t][w], rows m = b*TOT+t are contiguous in enc_out.
__global__ __launch_bounds__(256) void key_kernel(
    const _Float16* __restrict__ enc_out, const _Float16* __restrict__ Wk,
    float* __restrict__ KeyT) {
  const int wave = threadIdx.x >> 5, lane = threadIdx.x & 31;
  const int n0 = wave * 16;
  for (int mt = blockIdx.x; mt < (BB * TOT) / 16; mt += gridDim.x) {
    const _Float16* abase = enc_out + (long)mt * 16 * HH;
    const _Float16* wb = Wk + n0 * HH;
    v8f acc = {0.f, 0.f, 0.f, 0.f, 0.f, 0.f, 0.f, 0.f};
#pragma unroll
    for (int kk = 0; kk < 8; ++kk)
      acc = wmma_f16(load_frag(abase, HH, kk * 32), load_frag(wb, HH, kk * 32), acc);
    const int n = lane & 15, mo = (lane >> 4) * 8;
#pragma unroll
    for (int r = 0; r < 8; ++r)
      KeyT[((long)mt * 16 + mo + r) * WDIM + n0 + n] = acc[r];
  }
}

// ----------------------------------------------------------- decoder pieces -

__global__ __launch_bounds__(256) void init_beam(
    const _Float16* __restrict__ emb, const _Float16* __restrict__ enc_out,
    int* __restrict__ sched, float* __restrict__ lp, float* __restrict__ selm,
    float* __restrict__ mask, _Float16* __restrict__ din, _Float16* __restrict__ dh) {
  const int b = blockIdx.x;
  if (threadIdx.x == 0) { sched[b * 10] = -1; lp[b] = 0.f; }
  if (threadIdx.x < MAXS + 1) selm[b * (MAXS + 1) + threadIdx.x] = 0.f;
  for (int t = threadIdx.x; t < TOT; t += blockDim.x) mask[(long)b * TOT + t] = 0.f;
  if (threadIdx.x < DD) din[b * DD + threadIdx.x] = emb[DD + threadIdx.x];
  for (int j = threadIdx.x; j < HH; j += blockDim.x)
    dh[b * HH + j] = enc_out[((long)b * TOT + TSEQ) * HH + j];
}

// one decoder GRU step; grid = nbeams*8 blocks x 512 threads
__global__ __launch_bounds__(512) void gru_step(
    const _Float16* __restrict__ x, const _Float16* __restrict__ h,
    const _Float16* __restrict__ Wp, const float* __restrict__ bih,
    const float* __restrict__ bhh, _Float16* __restrict__ hnew) {
  __shared__ float grz[16][512];
  __shared__ float gin[16][HH];
  __shared__ float ghn[16][HH];
  const int m0 = blockIdx.x * 16;    // covers nbeams*BB rows
  const int wave = threadIdx.x >> 5, lane = threadIdx.x & 31;
  v16h ax[4];
#pragma unroll
  for (int kk = 0; kk < 4; ++kk) ax[kk] = load_frag(x + (long)m0 * DD, DD, kk * 32);
  v16h ah[8];
#pragma unroll
  for (int kk = 0; kk < 8; ++kk) ah[kk] = load_frag(h + (long)m0 * HH, HH, kk * 32);
#pragma unroll
  for (int i = 0; i < 3; ++i) {
    const int n0 = (wave * 3 + i) * 16;
    const _Float16* wb = Wp + (long)n0 * KW;
    v8f accI = {0.f, 0.f, 0.f, 0.f, 0.f, 0.f, 0.f, 0.f};
    v8f accH = {0.f, 0.f, 0.f, 0.f, 0.f, 0.f, 0.f, 0.f};
#pragma unroll
    for (int kk = 0; kk < 4; ++kk)
      accI = wmma_f16(ax[kk], load_frag(wb, KW, kk * 32), accI);
#pragma unroll
    for (int kk = 0; kk < 8; ++kk)
      accH = wmma_f16(ah[kk], load_frag(wb, KW, DD + kk * 32), accH);
    const int n = lane & 15, mo = (lane >> 4) * 8;
    if (n0 < 512) {
#pragma unroll
      for (int r = 0; r < 8; ++r) grz[mo + r][n0 + n] = accI[r] + accH[r];
    } else {
#pragma unroll
      for (int r = 0; r < 8; ++r) {
        gin[mo + r][n0 - 512 + n] = accI[r];
        ghn[mo + r][n0 - 512 + n] = accH[r];
      }
    }
  }
  __syncthreads();
  for (int i = threadIdx.x; i < 16 * HH; i += blockDim.x) {
    const int mm = i >> 8, j = i & 255;
    float r = sigm(grz[mm][j] + bih[j] + bhh[j]);
    float z = sigm(grz[mm][256 + j] + bih[256 + j] + bhh[256 + j]);
    float nv = fast_tanh(gin[mm][j] + bih[512 + j] + r * (ghn[mm][j] + bhh[512 + j]));
    float hp = (float)h[(long)(m0 + mm) * HH + j];
    hnew[(long)(m0 + mm) * HH + j] = (_Float16)((1.f - z) * nv + z * hp);
  }
}

// q[beam][b][w] = hn[beam][b][:] @ Wq^T ; grid = nbeams*8 blocks x 256
__global__ __launch_bounds__(256) void q_kernel(
    const _Float16* __restrict__ hn, const _Float16* __restrict__ Wq,
    float* __restrict__ q) {
  const int m0 = blockIdx.x * 16;
  const int wave = threadIdx.x >> 5, lane = threadIdx.x & 31;
  const int n0 = wave * 16;
  const _Float16* abase = hn + (long)m0 * HH;
  const _Float16* wb = Wq + n0 * HH;
  v8f acc = {0.f, 0.f, 0.f, 0.f, 0.f, 0.f, 0.f, 0.f};
#pragma unroll
  for (int kk = 0; kk < 8; ++kk)
    acc = wmma_f16(load_frag(abase, HH, kk * 32), load_frag(wb, HH, kk * 32), acc);
  const int n = lane & 15, mo = (lane >> 4) * 8;
#pragma unroll
  for (int r = 0; r < 8; ++r) q[(long)(m0 + mo + r) * WDIM + n0 + n] = acc[r];
}

// scores + softmax + top-4 per (beam,b); grid = nbeams*BB blocks x 256
__global__ __launch_bounds__(256) void attn_scores(
    const float* __restrict__ KeyT, const float* __restrict__ q,
    const float* __restrict__ wv, const float* __restrict__ mask,
    float* __restrict__ topk_lp, int* __restrict__ topk_i) {
  const int beam = blockIdx.x >> 7, b = blockIdx.x & 127;
  const int bb = beam * BB + b;
  __shared__ float sc[TOT];
  __shared__ float qs[WDIM];
  __shared__ float wvs[WDIM];
  __shared__ float sred[256];
  __shared__ int   sidx[256];
  if (threadIdx.x < WDIM) {
    qs[threadIdx.x]  = q[(long)bb * WDIM + threadIdx.x];
    wvs[threadIdx.x] = wv[threadIdx.x];
  }
  __syncthreads();
  for (int t = threadIdx.x; t < TOT; t += 256) {
    const float* kp = KeyT + ((long)b * TOT + t) * WDIM;
    float s = 0.f;
#pragma unroll 4
    for (int w = 0; w < WDIM; ++w) s += wvs[w] * fast_tanh(kp[w] + qs[w]);
    sc[t] = s - 1e7f * mask[(long)bb * TOT + t];
  }
  __syncthreads();
  // max
  float lm = -3.4e38f;
  for (int t = threadIdx.x; t < TOT; t += 256) lm = fmaxf(lm, sc[t]);
  sred[threadIdx.x] = lm;
  __syncthreads();
  for (int off = 128; off > 0; off >>= 1) {
    if ((int)threadIdx.x < off)
      sred[threadIdx.x] = fmaxf(sred[threadIdx.x], sred[threadIdx.x + off]);
    __syncthreads();
  }
  const float mx = sred[0];
  __syncthreads();
  // sum of exp
  float ls = 0.f;
  for (int t = threadIdx.x; t < TOT; t += 256) ls += __expf(sc[t] - mx);
  sred[threadIdx.x] = ls;
  __syncthreads();
  for (int off = 128; off > 0; off >>= 1) {
    if ((int)threadIdx.x < off) sred[threadIdx.x] += sred[threadIdx.x + off];
    __syncthreads();
  }
  const float lse = __logf(sred[0]);
  __syncthreads();
  // top-4 (stable: lower t wins ties)
  for (int k = 0; k < NBEAM; ++k) {
    float bv = -3.4e38f; int bi = TOT;
    for (int t = threadIdx.x; t < TOT; t += 256) {
      float v = sc[t];
      if (v > bv) { bv = v; bi = t; }
    }
    sred[threadIdx.x] = bv; sidx[threadIdx.x] = bi;
    __syncthreads();
    for (int off = 128; off > 0; off >>= 1) {
      if ((int)threadIdx.x < off) {
        float v2 = sred[threadIdx.x + off]; int i2 = sidx[threadIdx.x + off];
        if (v2 > sred[threadIdx.x] ||
            (v2 == sred[threadIdx.x] && i2 < sidx[threadIdx.x])) {
          sred[threadIdx.x] = v2; sidx[threadIdx.x] = i2;
        }
      }
      __syncthreads();
    }
    if (threadIdx.x == 0) {
      topk_lp[(long)bb * NBEAM + k] = sred[0] - mx - lse;
      topk_i[(long)bb * NBEAM + k]  = sidx[0];
      sc[sidx[0]] = -3.4e38f;
    }
    __syncthreads();
  }
}

// beam combine + gather; grid = BB blocks x 128 threads
__global__ __launch_bounds__(128) void beam_update(
    int ai, int nbeams,
    const int* __restrict__ sched, const float* __restrict__ lp,
    const float* __restrict__ selm, const float* __restrict__ mask,
    const _Float16* __restrict__ dec_hn, const _Float16* __restrict__ enc_in,
    const float* __restrict__ topk_lp, const int* __restrict__ topk_i,
    int* __restrict__ nsched, float* __restrict__ nlp, float* __restrict__ nselm,
    float* __restrict__ nmask, _Float16* __restrict__ ndin, _Float16* __restrict__ ndh) {
  const int b = blockIdx.x;
  __shared__ float cva[16];
  __shared__ int   cidx[16], cterm[16];
  __shared__ int   sel_src[NBEAM], sel_idx[NBEAM], sel_term[NBEAM];
  __shared__ float sel_lp[NBEAM];
  const int C = nbeams * NBEAM;
  if ((int)threadIdx.x < C) {
    const int c = threadIdx.x, bi = c >> 2, k = c & 3;
    const int ob = bi * BB + b;
    const int term = (sched[ob * 10 + ai] == 0);
    int idx = term ? 0 : topk_i[ob * NBEAM + k];
    if (ai == MAXS) idx = 0;
    const float l = topk_lp[ob * NBEAM + k];
    float np = lp[ob] + (k == 0 ? (term ? 0.f : l) : (l - (term ? 1e12f : 0.f)));
    cva[c] = np; cidx[c] = idx; cterm[c] = term;
  }
  __syncthreads();
  if (threadIdx.x == 0) {   // serial stable top-4 over <=16 scalars
    bool used[16];
    for (int c = 0; c < 16; ++c) used[c] = false;
    for (int s = 0; s < NBEAM; ++s) {
      float best = -3.4e38f; int bc = 0;
      for (int c = 0; c < C; ++c)
        if (!used[c] && cva[c] > best) { best = cva[c]; bc = c; }
      used[bc] = true;
      sel_src[s] = bc >> 2; sel_idx[s] = cidx[bc];
      sel_term[s] = cterm[bc]; sel_lp[s] = best;
    }
  }
  __syncthreads();
  for (int s = 0; s < NBEAM; ++s) {
    const int src = sel_src[s], idx = sel_idx[s], term = sel_term[s];
    const long ob = (long)src * BB + b;
    const long nb = (long)s * BB + b;
    if ((int)threadIdx.x <= ai) nsched[nb * 10 + threadIdx.x] = sched[ob * 10 + threadIdx.x];
    if (threadIdx.x == 0) { nsched[nb * 10 + ai + 1] = idx; nlp[nb] = sel_lp[s]; }
    if (threadIdx.x < MAXS + 1) {
      float v = selm[ob * (MAXS + 1) + threadIdx.x];
      if ((int)threadIdx.x == ai && !term) v = 1.f;
      nselm[nb * (MAXS + 1) + threadIdx.x] = v;
    }
    for (int t = threadIdx.x; t < TOT; t += 128) {
      float v = mask[ob * TOT + t];
      if (t == idx) v = 1.f;
      nmask[nb * TOT + t] = v;
    }
    if (threadIdx.x < DD)
      ndin[nb * DD + threadIdx.x] = enc_in[((long)b * TOT + idx) * DD + threadIdx.x];
    for (int j = threadIdx.x; j < HH; j += 128) ndh[nb * HH + j] = dec_hn[ob * HH + j];
    __syncthreads();
  }
}

__global__ void write_out(const float* __restrict__ lp, const float* __restrict__ mask,
                          float* __restrict__ out) {
  const long total = (long)BB * (TOT + 1);
  for (long i = blockIdx.x * blockDim.x + threadIdx.x; i < total;
       i += (long)gridDim.x * blockDim.x) {
    if (i < BB) out[i] = lp[i];                  // beam0 logprob [B,1]
    else        out[i] = mask[i - BB];           // beam0 mask [B,TOT]
  }
}

// ------------------------------------------------------------------- host ---

extern "C" void kernel_launch(void* const* d_in, const int* in_sizes, int n_in,
                              void* d_out, int out_size, void* d_ws, size_t ws_size,
                              hipStream_t stream) {
  (void)in_sizes; (void)n_in; (void)out_size; (void)ws_size;
  const float* cm   = (const float*)d_in[0];
  const float* emb  = (const float*)d_in[1];
  const float* linW = (const float*)d_in[2];
  const float* linb = (const float*)d_in[3];
  const float* eWih = (const float*)d_in[4];
  const float* eWhh = (const float*)d_in[5];
  const float* ebih = (const float*)d_in[6];
  const float* ebhh = (const float*)d_in[7];
  const float* dWih = (const float*)d_in[8];
  const float* dWhh = (const float*)d_in[9];
  const float* dbih = (const float*)d_in[10];
  const float* dbhh = (const float*)d_in[11];
  const float* Wk   = (const float*)d_in[12];
  const float* Wq   = (const float*)d_in[13];
  const float* wv   = (const float*)d_in[14];
  float* out = (float*)d_out;

  char* ws = (char*)d_ws;
  size_t off = 0;
  auto alloc = [&](size_t bytes) -> void* {
    void* p = ws + off;
    off = (off + bytes + 255) & ~(size_t)255;
    return p;
  };

  _Float16* enc_in_h  = (_Float16*)alloc((size_t)BB * TOT * DD * 2);
  _Float16* enc_out_h = (_Float16*)alloc((size_t)BB * TOT * HH * 2);
  float*    KeyT      = (float*)   alloc((size_t)BB * TOT * WDIM * 4);
  _Float16* wl_h      = (_Float16*)alloc((size_t)DD * FF * 2);
  _Float16* wenc_h    = (_Float16*)alloc((size_t)GG * KW * 2);
  _Float16* wdec_h    = (_Float16*)alloc((size_t)GG * KW * 2);
  _Float16* wk_h      = (_Float16*)alloc((size_t)WDIM * HH * 2);
  _Float16* wq_h      = (_Float16*)alloc((size_t)WDIM * HH * 2);
  _Float16* emb_h     = (_Float16*)alloc((size_t)2 * DD * 2);
  _Float16* dec_hn    = (_Float16*)alloc((size_t)NBEAM * BB * HH * 2);
  float*    qbuf      = (float*)   alloc((size_t)NBEAM * BB * WDIM * 4);
  float*    topk_lp   = (float*)   alloc((size_t)NBEAM * BB * NBEAM * 4);
  int*      topk_i    = (int*)     alloc((size_t)NBEAM * BB * NBEAM * 4);

  struct BeamBufs { int* sched; float* lp; float* selm; float* mask;
                    _Float16* din; _Float16* dh; };
  BeamBufs st[2];
  for (int s = 0; s < 2; ++s) {
    st[s].sched = (int*)     alloc((size_t)NBEAM * BB * 10 * 4);
    st[s].lp    = (float*)   alloc((size_t)NBEAM * BB * 4);
    st[s].selm  = (float*)   alloc((size_t)NBEAM * BB * (MAXS + 1) * 4);
    st[s].mask  = (float*)   alloc((size_t)NBEAM * BB * TOT * 4);
    st[s].din   = (_Float16*)alloc((size_t)NBEAM * BB * DD * 2);
    st[s].dh    = (_Float16*)alloc((size_t)NBEAM * BB * HH * 2);
  }

  // weights -> f16
  cvt_f16<<<32, 256, 0, stream>>>(wl_h, linW, DD * FF);
  cvt_f16<<<128, 256, 0, stream>>>(wk_h, Wk, WDIM * HH);
  cvt_f16<<<128, 256, 0, stream>>>(wq_h, Wq, WDIM * HH);
  cvt_f16<<<1, 256, 0, stream>>>(emb_h, emb, 2 * DD);
  pack_gru<<<1152, 256, 0, stream>>>(wenc_h, eWih, eWhh);
  pack_gru<<<1152, 256, 0, stream>>>(wdec_h, dWih, dWhh);

  // encoder input: BOS + projection
  init_bos<<<64, 256, 0, stream>>>(emb_h, enc_in_h);
  proj_kernel<<<2048, 256, 0, stream>>>(cm, wl_h, linb, enc_in_h);

  // persistent GRU scan (8 WGPs, batch-split, TDM-staged x tiles)
  encoder_scan<<<8, 512, 0, stream>>>(enc_in_h, wenc_h, ebih, ebhh, enc_out_h);

  // Key projection
  key_kernel<<<2048, 256, 0, stream>>>(enc_out_h, wk_h, KeyT);

  // beam 0 init
  init_beam<<<BB, 256, 0, stream>>>(emb_h, enc_out_h, st[0].sched, st[0].lp,
                                    st[0].selm, st[0].mask, st[0].din, st[0].dh);

  BeamBufs* cur = &st[0];
  BeamBufs* nxt = &st[1];
  for (int ai = 0; ai <= MAXS; ++ai) {
    const int nb = (ai == 0) ? 1 : NBEAM;
    gru_step<<<nb * 8, 512, 0, stream>>>(cur->din, cur->dh, wdec_h, dbih, dbhh, dec_hn);
    q_kernel<<<nb * 8, 256, 0, stream>>>(dec_hn, wq_h, qbuf);
    attn_scores<<<nb * BB, 256, 0, stream>>>(KeyT, qbuf, wv, cur->mask, topk_lp, topk_i);
    beam_update<<<BB, 128, 0, stream>>>(ai, nb, cur->sched, cur->lp, cur->selm,
                                        cur->mask, dec_hn, enc_in_h, topk_lp, topk_i,
                                        nxt->sched, nxt->lp, nxt->selm, nxt->mask,
                                        nxt->din, nxt->dh);
    BeamBufs* tmp = cur; cur = nxt; nxt = tmp;
  }

  write_out<<<514, 256, 0, stream>>>(cur->lp, cur->mask, out);
}
